// ChaosAttention_30769145709114
// MI455X (gfx1250) — matrix operand order repaired and checked by reading, not compiled, not run on hardware
//
#include <hip/hip_runtime.h>

// ---------------------------------------------------------------------------
// ChaosAttention for MI455X (gfx1250): bf16 WMMA pipeline, software-pipelined
//   B=2, L=2048, D=1024, H=16, HD=64
// ---------------------------------------------------------------------------

typedef __bf16 v16bf __attribute__((ext_vector_type(16)));
typedef float  v8f   __attribute__((ext_vector_type(8)));
typedef unsigned int u32x4 __attribute__((ext_vector_type(4)));
typedef unsigned short u16;

union Frag {
    u32x4 u[2];   // 32 bytes
    v16bf v;      // 16 x bf16
};

__device__ __forceinline__ u16 f2bf(float f) {
    unsigned u = __builtin_bit_cast(unsigned, f);
    u += 0x7FFFu + ((u >> 16) & 1u);   // round-to-nearest-even
    return (u16)(u >> 16);
}

__device__ __forceinline__ v8f wmma_bf16(v16bf a, v16bf b, v8f c) {
    return __builtin_amdgcn_wmma_f32_16x16x32_bf16(
        false, a, false, b, (short)0, c, false, false);
}

// ------------------------- conversion kernels ------------------------------

__global__ void cvt_bf16_kernel(const float* __restrict__ in,
                                u16* __restrict__ out, int n) {
    int i = blockIdx.x * blockDim.x + threadIdx.x;
    if (i < n) out[i] = f2bf(in[i]);
}

// in: [1024 x 1024] row-major (K x N) -> out: [N x K] bf16 (N-major for B-frags)
__global__ void cvtT_kernel(const float* __restrict__ in, u16* __restrict__ out) {
    int i = blockIdx.x * blockDim.x + threadIdx.x;   // 0..1048575
    int krow = i >> 10, ncol = i & 1023;
    out[(size_t)ncol * 1024 + krow] = f2bf(in[i]);
}

// ------------------------- chaos / gate kernel -----------------------------
__global__ __launch_bounds__(128)
void chaos_kernel(const float* __restrict__ ci, const float* __restrict__ Wc,
                  const float* __restrict__ bc, const float* __restrict__ Wg,
                  const float* __restrict__ bg, u16* __restrict__ cq,
                  float* __restrict__ gate) {
    int wave = threadIdx.x >> 5, lane = threadIdx.x & 31;
    int tok = blockIdx.x * 4 + wave;          // 0..4095
    int b = tok >> 11, l = tok & 2047;

    float xv = ci[tok * 3 + 0], yv = ci[tok * 3 + 1], zv = ci[tok * 3 + 2];
    float ax = 0.f, ay = 0.f, az = 0.f;
#pragma unroll
    for (int i = 0; i < 10; ++i) {
        float dx = 10.f * (yv - xv);
        float dy = xv * (28.f - zv) - yv;
        float dz = xv * yv - (8.f / 3.f) * zv;
        xv += 0.01f * dx; yv += 0.01f * dy; zv += 0.01f * dz;
        ax += xv; ay += yv; az += zv;
    }
    ax *= 0.1f; ay *= 0.1f; az *= 0.1f;

    float ga = 0.f;
    for (int d = lane; d < 1024; d += 32) {
        float f = ax * Wc[d] + ay * Wc[1024 + d] + az * Wc[2048 + d] + bc[d];
        ga += f * Wg[d];
        int h = d >> 6, hd = d & 63;
        cq[((size_t)(b * 16 + h) * 2048 + l) * 64 + hd] = f2bf(f);
    }
    ga += __shfl_xor(ga, 1, 32);
    ga += __shfl_xor(ga, 2, 32);
    ga += __shfl_xor(ga, 4, 32);
    ga += __shfl_xor(ga, 8, 32);
    ga += __shfl_xor(ga, 16, 32);
    if (lane == 0) gate[tok] = 1.f / (1.f + __expf(-(ga + bg[0])));
}

// ------------------------- WMMA GEMM (32x64 tile, double-buffered) ---------
// C[4096,1024] = A[4096,1024](bf16) @ W (Wt stored [N,K] bf16) + bias.
// mode 0: f32 plain out. mode 1: bf16 [B,H,L,HD]. mode 2: bf16 [B,H,HD,L].
__global__ __launch_bounds__(128)
void gemm32(const u16* __restrict__ A, const u16* __restrict__ Wt,
            const float* __restrict__ bias, float* __restrict__ outF,
            u16* __restrict__ outB, int mode) {
    constexpr int K = 1024, N = 1024;
    int wave = threadIdx.x >> 5, lane = threadIdx.x & 31;
    int n = lane & 15, kh = lane >> 4;
    int t  = blockIdx.x * 4 + wave;   // 0..2047
    int mt = t & 127;                 // 128 row tiles of 32
    int nb = t >> 7;                  // 16 col blocks of 64

    const u16* ar0 = A + (size_t)(mt * 32 + n) * K;
    const u16* ar1 = ar0 + (size_t)16 * K;

    auto loadA = [&](Frag& f0, Frag& f1, int k0) {
        f0.u[0] = *(const u32x4*)(ar0 + k0 + kh * 8);
        f0.u[1] = *(const u32x4*)(ar0 + k0 + 16 + kh * 8);
        f1.u[0] = *(const u32x4*)(ar1 + k0 + kh * 8);
        f1.u[1] = *(const u32x4*)(ar1 + k0 + 16 + kh * 8);
    };
    auto loadB = [&](Frag (&fb)[4], int k0) {
#pragma unroll
        for (int tt = 0; tt < 4; ++tt) {
            const u16* w = Wt + (size_t)(nb * 64 + tt * 16 + n) * K + k0 + kh * 8;
            fb[tt].u[0] = *(const u32x4*)(w);
            fb[tt].u[1] = *(const u32x4*)(w + 16);
        }
    };

    v8f acc[8] = {v8f{}, v8f{}, v8f{}, v8f{}, v8f{}, v8f{}, v8f{}, v8f{}};

    Frag a0A, a1A, bA[4];         // set A (even k-steps)
    Frag a0B, a1B, bB[4];         // set B (odd k-steps)
    loadA(a0A, a1A, 0);
    loadB(bA, 0);

    for (int k0 = 0; k0 < K; k0 += 64) {
        // prefetch odd step while even step computes
        loadA(a0B, a1B, k0 + 32);
        loadB(bB, k0 + 32);
#pragma unroll
        for (int tt = 0; tt < 4; ++tt) {
            acc[tt]     = wmma_bf16(a0A.v, bA[tt].v, acc[tt]);
            acc[4 + tt] = wmma_bf16(a1A.v, bA[tt].v, acc[4 + tt]);
        }
        // prefetch next even step while odd step computes
        if (k0 + 64 < K) {
            loadA(a0A, a1A, k0 + 64);
            loadB(bA, k0 + 64);
        }
#pragma unroll
        for (int tt = 0; tt < 4; ++tt) {
            acc[tt]     = wmma_bf16(a0B.v, bB[tt].v, acc[tt]);
            acc[4 + tt] = wmma_bf16(a1B.v, bB[tt].v, acc[4 + tt]);
        }
    }

#pragma unroll
    for (int half = 0; half < 2; ++half) {
#pragma unroll
        for (int tt = 0; tt < 4; ++tt) {
            int col = nb * 64 + tt * 16 + n;
            float bvv = bias[col];
#pragma unroll
            for (int r = 0; r < 8; ++r) {
                int row = mt * 32 + half * 16 + r + 8 * kh;
                float val = acc[half * 4 + tt][r] + bvv;
                if (mode == 0) {
                    outF[(size_t)row * N + col] = val;
                } else {
                    int bi = row >> 11, li = row & 2047;
                    int h = col >> 6, d = col & 63;
                    if (mode == 1)
                        outB[((size_t)(bi * 16 + h) * 2048 + li) * 64 + d] = f2bf(val);
                    else
                        outB[((size_t)(bi * 16 + h) * 64 + d) * 2048 + li] = f2bf(val);
                }
            }
        }
    }
}

// ------------------------- fused flash attention (pipelined) ---------------
// One wave per (b, h, 16-row q tile). Online softmax over 2048 keys, 32/iter.
// scores = (q.k + 0.1*gate[key]*(cq.k)) / 8 ; mask==0 -> -inf ; O = softmax@V.
__global__ __launch_bounds__(128)
void attn_kernel(const u16* __restrict__ q, const u16* __restrict__ kmat,
                 const u16* __restrict__ vT, const u16* __restrict__ cq,
                 const float* __restrict__ gate, const int* __restrict__ mask,
                 u16* __restrict__ ao) {
    __shared__ u16 plds[4 * 16 * 32];          // 1KB P-scratch per wave
    int wave = threadIdx.x >> 5, lane = threadIdx.x & 31;
    int n = lane & 15, kh = lane >> 4;
    int wbase = wave * 512;

    int tile = blockIdx.x * 4 + wave;   // 0..4095
    int qt = tile & 127;                // q tile within L
    int bh = tile >> 7;                 // b*16 + h
    int b  = bh >> 4;

    const u16* qbase  = q    + ((size_t)bh * 2048 + qt * 16) * 64;
    const u16* kbp    = kmat + (size_t)bh * 2048 * 64;
    const u16* vbase  = vT   + (size_t)bh * 64 * 2048;
    const u16* cqbase = cq   + ((size_t)bh * 2048 + qt * 16) * 64;
    const float* gb   = gate + b * 2048;
    const int* mb     = mask + b * 2048;

    // Q / CQ A-fragments (K = d, 0..31 and 32..63)
    Frag a0, a1, ca0, ca1;
    {
        const u16* qr = qbase + n * 64;
        a0.u[0] = *(const u32x4*)(qr + kh * 8);
        a0.u[1] = *(const u32x4*)(qr + 16 + kh * 8);
        a1.u[0] = *(const u32x4*)(qr + 32 + kh * 8);
        a1.u[1] = *(const u32x4*)(qr + 48 + kh * 8);
        const u16* cr = cqbase + n * 64;
        ca0.u[0] = *(const u32x4*)(cr + kh * 8);
        ca0.u[1] = *(const u32x4*)(cr + 16 + kh * 8);
        ca1.u[0] = *(const u32x4*)(cr + 32 + kh * 8);
        ca1.u[1] = *(const u32x4*)(cr + 48 + kh * 8);
    }

    // K fragments, double-duty registers: reloaded for tile kb+32 right after
    // their last use for tile kb (keeps loop-top WMMAs wait-free).
    Frag k00, k01, k10, k11;
    auto loadK = [&](int kb) {
        const u16* kr0 = kbp + (size_t)(kb + n) * 64;
        const u16* kr1 = kr0 + 16 * 64;
        k00.u[0] = *(const u32x4*)(kr0 + kh * 8);
        k00.u[1] = *(const u32x4*)(kr0 + 16 + kh * 8);
        k01.u[0] = *(const u32x4*)(kr0 + 32 + kh * 8);
        k01.u[1] = *(const u32x4*)(kr0 + 48 + kh * 8);
        k10.u[0] = *(const u32x4*)(kr1 + kh * 8);
        k10.u[1] = *(const u32x4*)(kr1 + 16 + kh * 8);
        k11.u[0] = *(const u32x4*)(kr1 + 32 + kh * 8);
        k11.u[1] = *(const u32x4*)(kr1 + 48 + kh * 8);
    };
    loadK(0);
    float g0 = gb[n], g1 = gb[16 + n];
    int   m0 = mb[n], m1 = mb[16 + n];

    v8f o0 = {}, o1 = {}, o2 = {}, o3 = {};
    float mrow[8], lrow[8];
#pragma unroll
    for (int r = 0; r < 8; ++r) { mrow[r] = -1e30f; lrow[r] = 0.f; }

    for (int kb = 0; kb < 2048; kb += 32) {
        // V fragments issued early: latency hidden behind score WMMAs+softmax
        Frag vf[4];
#pragma unroll
        for (int dc = 0; dc < 4; ++dc) {
            const u16* vr = vbase + (size_t)(dc * 16 + n) * 2048 + kb + kh * 8;
            vf[dc].u[0] = *(const u32x4*)(vr);
            vf[dc].u[1] = *(const u32x4*)(vr + 16);
        }

        v8f z = {};
        v8f s0 = wmma_bf16(a0.v, k00.v, z);  s0 = wmma_bf16(a1.v, k01.v, s0);
        v8f s1 = wmma_bf16(a0.v, k10.v, z);  s1 = wmma_bf16(a1.v, k11.v, s1);
        v8f c0 = wmma_bf16(ca0.v, k00.v, z); c0 = wmma_bf16(ca1.v, k01.v, c0);
        v8f c1 = wmma_bf16(ca0.v, k10.v, z); c1 = wmma_bf16(ca1.v, k11.v, c1);

        float gg0 = g0, gg1 = g1;
        int   mm0 = m0, mm1 = m1;
        // prefetch next tile's K/gate/mask into the now-free registers
        if (kb + 32 < 2048) {
            loadK(kb + 32);
            g0 = gb[kb + 32 + n]; g1 = gb[kb + 48 + n];
            m0 = mb[kb + 32 + n]; m1 = mb[kb + 48 + n];
        }

#pragma unroll
        for (int r = 0; r < 8; ++r) {
            float f0 = (s0[r] + 0.1f * gg0 * c0[r]) * 0.125f;
            float f1 = (s1[r] + 0.1f * gg1 * c1[r]) * 0.125f;
            if (mm0 == 0) f0 = -1e30f;
            if (mm1 == 0) f1 = -1e30f;
            float t = fmaxf(f0, f1);
            t = fmaxf(t, __shfl_xor(t, 1, 16));
            t = fmaxf(t, __shfl_xor(t, 2, 16));
            t = fmaxf(t, __shfl_xor(t, 4, 16));
            t = fmaxf(t, __shfl_xor(t, 8, 16));
            float mnew = fmaxf(mrow[r], t);
            float corr = __expf(mrow[r] - mnew);
            float p0 = __expf(f0 - mnew);
            float p1 = __expf(f1 - mnew);
            float rs = p0 + p1;
            rs += __shfl_xor(rs, 1, 16);
            rs += __shfl_xor(rs, 2, 16);
            rs += __shfl_xor(rs, 4, 16);
            rs += __shfl_xor(rs, 8, 16);
            lrow[r] = lrow[r] * corr + rs;
            mrow[r] = mnew;
            o0[r] *= corr; o1[r] *= corr; o2[r] *= corr; o3[r] *= corr;
            plds[wbase + (r + 8 * kh) * 32 + n]      = f2bf(p0);
            plds[wbase + (r + 8 * kh) * 32 + 16 + n] = f2bf(p1);
        }
        asm volatile("s_wait_dscnt 0" ::: "memory");

        // P A-fragment (16 rows x 32 keys) from LDS
        Frag pa;
        const u16* pl = &plds[wbase + n * 32 + kh * 8];
        pa.u[0] = *(const u32x4*)(pl);
        pa.u[1] = *(const u32x4*)(pl + 16);

        // O += P @ V  (V fragments landed long ago)
        o0 = wmma_bf16(pa.v, vf[0].v, o0);
        o1 = wmma_bf16(pa.v, vf[1].v, o1);
        o2 = wmma_bf16(pa.v, vf[2].v, o2);
        o3 = wmma_bf16(pa.v, vf[3].v, o3);
    }

    int h = bh & 15;
#pragma unroll
    for (int r = 0; r < 8; ++r) {
        float inv = 1.0f / lrow[r];
        int row = b * 2048 + qt * 16 + r + 8 * kh;
        size_t base = (size_t)row * 1024 + h * 64 + n;
        ao[base +  0] = f2bf(o0[r] * inv);
        ao[base + 16] = f2bf(o1[r] * inv);
        ao[base + 32] = f2bf(o2[r] * inv);
        ao[base + 48] = f2bf(o3[r] * inv);
    }
}

// ------------------------- launch ------------------------------------------

extern "C" void kernel_launch(void* const* d_in, const int* in_sizes, int n_in,
                              void* d_out, int out_size, void* d_ws, size_t ws_size,
                              hipStream_t stream) {
    const float* x    = (const float*)d_in[0];
    const int*   mask = (const int*)  d_in[1];
    const float* ci   = (const float*)d_in[2];
    const float* Wq   = (const float*)d_in[3];
    const float* bq   = (const float*)d_in[4];
    const float* Wk   = (const float*)d_in[5];
    const float* bk   = (const float*)d_in[6];
    const float* Wv   = (const float*)d_in[7];
    const float* bv   = (const float*)d_in[8];
    const float* Wo   = (const float*)d_in[9];
    const float* bo   = (const float*)d_in[10];
    const float* Wc   = (const float*)d_in[11];
    const float* bc   = (const float*)d_in[12];
    const float* Wg   = (const float*)d_in[13];
    const float* bg   = (const float*)d_in[14];

    char* ws = (char*)d_ws;
    const size_t TOK = 4096, D = 1024;
    u16*   xb   = (u16*)(ws + 0);                          //  8 MB  x bf16
    u16*   wt   = (u16*)(ws + 8388608);                    //  2 MB  W^T bf16 (reused)
    u16*   qb   = (u16*)(ws + 10485760);                   //  8 MB  Q  [B,H,L,HD]
    u16*   kbuf = (u16*)(ws + 18874368);                   //  8 MB  K  [B,H,L,HD]
    u16*   vb_  = (u16*)(ws + 27262976);                   //  8 MB  V^T [B,H,HD,L]
    u16*   cqb  = (u16*)(ws + 35651584);                   //  8 MB  CQ [B,H,L,HD]
    float* gb   = (float*)(ws + 44040192);                 // 16 KB  gate
    u16*   ao   = (u16*)(ws + 44056576);                   //  8 MB  attn out bf16

    (void)in_sizes; (void)n_in; (void)out_size; (void)ws_size;

    // 1. x -> bf16
    cvt_bf16_kernel<<<(int)((TOK * D + 255) / 256), 256, 0, stream>>>(x, xb, (int)(TOK * D));
    // 2. chaos features (bf16, head layout) + gate
    chaos_kernel<<<1024, 128, 0, stream>>>(ci, Wc, bc, Wg, bg, cqb, gb);
    // 3. Q projection
    cvtT_kernel<<<4096, 256, 0, stream>>>(Wq, wt);
    gemm32<<<512, 128, 0, stream>>>(xb, wt, bq, nullptr, qb, 1);
    // 4. K projection
    cvtT_kernel<<<4096, 256, 0, stream>>>(Wk, wt);
    gemm32<<<512, 128, 0, stream>>>(xb, wt, bk, nullptr, kbuf, 1);
    // 5. V projection (stored transposed [B,H,HD,L])
    cvtT_kernel<<<4096, 256, 0, stream>>>(Wv, wt);
    gemm32<<<512, 128, 0, stream>>>(xb, wt, bv, nullptr, vb_, 2);
    // 6. fused chaos-gated flash attention
    attn_kernel<<<1024, 128, 0, stream>>>(qb, kbuf, vb_, cqb, gb, mask, ao);
    // 7. output projection -> f32 d_out
    cvtT_kernel<<<4096, 256, 0, stream>>>(Wo, wt);
    gemm32<<<512, 128, 0, stream>>>(ao, wt, bo, (float*)d_out, nullptr, 0);
}
